// GAT_1236950581715
// MI455X (gfx1250) — matrix operand order repaired and checked by reading, not compiled, not run on hardware
//
#include <hip/hip_runtime.h>
#include <cstdint>
#include <cstddef>

typedef __attribute__((ext_vector_type(16))) _Float16 v16h;
typedef __attribute__((ext_vector_type(8)))  _Float16 v8h;
typedef __attribute__((ext_vector_type(8)))  float    v8f;
typedef __attribute__((ext_vector_type(4)))  float    v4f;
typedef __attribute__((ext_vector_type(4)))  int      v4i;

// pointee-qualified types for the async-copy builtin (sig learned from clang
// diagnostic: arg0 = global int4*, arg1 = LDS int4*, then imm offset, imm cpol)
typedef __attribute__((address_space(1))) v4i gv4i;
typedef __attribute__((address_space(3))) v4i sv4i;

#ifndef __has_builtin
#define __has_builtin(x) 0
#endif
#if __has_builtin(__builtin_amdgcn_global_load_async_to_lds_b128) && \
    __has_builtin(__builtin_amdgcn_s_wait_asynccnt)
#define HAVE_ASYNC_LDS 1
#else
#define HAVE_ASYNC_LDS 0
#endif

// ---------------------------------------------------------------- helpers
__device__ __forceinline__ float leaky02(float x) { return x > 0.f ? x : 0.2f * x; }

// Correct float atomic max via sign-split int atomics (valid with -inf init).
__device__ __forceinline__ void atomic_max_f32(float* addr, float val) {
  if (val >= 0.f) atomicMax((int*)addr, __float_as_int(val));
  else            atomicMin((unsigned int*)addr, __float_as_uint(val));
}

__global__ void fill_f32(float* __restrict__ p, float v, size_t n) {
  size_t i = (size_t)blockIdx.x * blockDim.x + threadIdx.x;
  if (i < n) p[i] = v;
}

// ---------------------------------------------------------------- GEMM (WMMA)
// C[M,NCOLS] = A[M,K] * B[K,NCOLS]; fp32 in/out, f16 WMMA, f32 accumulate.
// Block = NCOLS/16 waves; wave w owns column tile w. Persistent over row tiles.
// B is staged once per block in LDS, f16 + transposed -> B fragment = 2x ds_load_b128.
// A strip (16 x K f32) is staged per row tile (async DMA to LDS when available),
// shared by all waves -> 8x less global A traffic than per-wave loads.
#define APAD 4   // f32 pad per sA row
#define BPAD 8   // f16 pad per sBT row

template <int K, int NCOLS>
__global__ void __launch_bounds__(NCOLS * 2)
gemm_wmma_f16(const float* __restrict__ A, const float* __restrict__ B,
              float* __restrict__ C, int M) {
  constexpr int NW  = NCOLS / 16;   // waves per block
  constexpr int TPB = NW * 32;      // threads per block
  __shared__ float    sA[16 * (K + APAD)];
  __shared__ _Float16 sBT[NCOLS * (K + BPAD)];

  const int tid  = threadIdx.x;
  const int wave = tid >> 5;
  const int lane = tid & 31;
  const int m    = lane & 15;
  const int kh   = lane >> 4;       // lane half selects K phase (ISA 7.12.2)
  const int col0 = wave * 16;

  // ---- stage B once per block: f16, transposed (column-major rows) ----
  for (int idx = tid; idx < K * NCOLS; idx += TPB) {
    int k = idx / NCOLS, n = idx - k * NCOLS;          // coalesced global read
    sBT[n * (K + BPAD) + k] = (_Float16)B[idx];
  }

  const int rowTiles = (M + 15) >> 4;
  for (int rt = blockIdx.x; rt < rowTiles; rt += gridDim.x) {
    const int row0 = rt * 16;
    __syncthreads();   // covers B fill (1st iter) and sA reuse (later iters)

    // ---- stage A strip: 16 rows x K f32 ----
#if HAVE_ASYNC_LDS
    for (int c = tid; c < (16 * K) / 4; c += TPB) {
      int mm = c / (K / 4), kk = (c - mm * (K / 4)) * 4;
      int row = row0 + mm; if (row >= M) row = M - 1;
      __builtin_amdgcn_global_load_async_to_lds_b128(
          (gv4i*)(A + (size_t)row * K + kk),
          (sv4i*)&sA[mm * (K + APAD) + kk],
          0, 0);
    }
    __builtin_amdgcn_s_wait_asynccnt(0);
#else
    for (int c = tid; c < (16 * K) / 4; c += TPB) {
      int mm = c / (K / 4), kk = (c - mm * (K / 4)) * 4;
      int row = row0 + mm; if (row >= M) row = M - 1;
      float4 v = *(const float4*)(A + (size_t)row * K + kk);
      *(float4*)&sA[mm * (K + APAD) + kk] = v;
    }
#endif
    __syncthreads();

    v8f acc = {};
    #pragma unroll
    for (int k0 = 0; k0 < K; k0 += 32) {
      // A fragment: lane m, K runs {k0+kh*8 ..+7} and {k0+16+kh*8 ..+7}
      const float* ar = &sA[m * (K + APAD)];
      v4f a0 = *(const v4f*)&ar[k0 + kh * 8];
      v4f a1 = *(const v4f*)&ar[k0 + kh * 8 + 4];
      v4f a2 = *(const v4f*)&ar[k0 + 16 + kh * 8];
      v4f a3 = *(const v4f*)&ar[k0 + 16 + kh * 8 + 4];
      v16h a;
      #pragma unroll
      for (int i = 0; i < 4; ++i) {
        a[i]      = (_Float16)a0[i];
        a[i + 4]  = (_Float16)a1[i];
        a[i + 8]  = (_Float16)a2[i];
        a[i + 12] = (_Float16)a3[i];
      }
      // B fragment: column col0+m, K runs {k0+kh*8 ..+7} and {k0+16+kh*8 ..+7}
      const _Float16* br = &sBT[(col0 + m) * (K + BPAD) + k0 + kh * 8];
      v8h b0 = *(const v8h*)br;
      v8h b1 = *(const v8h*)(br + 16);
      v16h b;
      #pragma unroll
      for (int i = 0; i < 8; ++i) { b[i] = b0[i]; b[i + 8] = b1[i]; }

      acc = __builtin_amdgcn_wmma_f32_16x16x32_f16(
          /*neg_a=*/false, a, /*neg_b=*/false, b,
          /*c_mod=*/(short)0, acc, /*reuse_a=*/false, /*reuse_b=*/false);
    }

    // C layout: VGPR r -> M = r + kh*8, N = col0 + m
    const int cn = col0 + m;
    #pragma unroll
    for (int r = 0; r < 8; ++r) {
      int cm = row0 + kh * 8 + r; if (cm >= M) cm = M - 1;
      C[(size_t)cm * NCOLS + cn] = acc[r];
    }
  }
}

// ---------------------------------------------------------------- attention dots
// a_src[n,h] = <xl[n,h,:], att_src[h,:]>, a_dst likewise. One thread per (n,h).
__global__ void attn_coef(const float* __restrict__ xl,
                          const float* __restrict__ att_src,
                          const float* __restrict__ att_dst,
                          float* __restrict__ a_src, float* __restrict__ a_dst,
                          int Nn, int H, int C) {
  int i = blockIdx.x * blockDim.x + threadIdx.x;
  if (i >= Nn * H) return;
  int n = i / H, h = i - n * H;
  const float4* xr = (const float4*)(xl + (size_t)n * H * C + (size_t)h * C);
  const float4* as = (const float4*)(att_src + (size_t)h * C);
  const float4* ad = (const float4*)(att_dst + (size_t)h * C);
  float s = 0.f, d = 0.f;
  for (int c = 0; c < C / 4; ++c) {
    float4 v = xr[c], va = as[c], vd = ad[c];
    s += v.x * va.x + v.y * va.y + v.z * va.z + v.w * va.w;
    d += v.x * vd.x + v.y * vd.y + v.z * vd.z + v.w * vd.w;
  }
  a_src[i] = s; a_dst[i] = d;
}

// ---------------------------------------------------------------- edge passes
// Work item = (edge, head). Edges 0..E-1 from edge_index; E..E+Nn-1 self-loops.
__device__ __forceinline__ void edge_sd(const int* ei, int E, int e, int& s, int& d) {
  if (e < E) { s = ei[e]; d = ei[E + e]; } else { s = d = e - E; }
}

__global__ void edge_max(const int* __restrict__ ei, int E, int Nn,
                         const float* __restrict__ a_src, const float* __restrict__ a_dst,
                         float* __restrict__ emax, int H) {
  int i = blockIdx.x * blockDim.x + threadIdx.x;
  int total = (E + Nn) * H;
  if (i >= total) return;
  int e = i / H, h = i - e * H, s, d;
  edge_sd(ei, E, e, s, d);
  float v = leaky02(a_src[s * H + h] + a_dst[d * H + h]);
  atomic_max_f32(&emax[d * H + h], v);
}

__global__ void edge_exp(const int* __restrict__ ei, int E, int Nn,
                         const float* __restrict__ a_src, const float* __restrict__ a_dst,
                         const float* __restrict__ emax,
                         float* __restrict__ eexp, float* __restrict__ denom, int H) {
  int i = blockIdx.x * blockDim.x + threadIdx.x;
  int total = (E + Nn) * H;
  if (i >= total) return;
  int e = i / H, h = i - e * H, s, d;
  edge_sd(ei, E, e, s, d);
  float v  = leaky02(a_src[s * H + h] + a_dst[d * H + h]);
  float ex = __expf(v - emax[d * H + h]);
  eexp[i] = ex;
  atomicAdd(&denom[d * H + h], ex);
}

// One thread per (edge, 16-float chunk): gather xl[src], scatter-add into out[dst].
__global__ void edge_msg(const int* __restrict__ ei, int E, int Nn,
                         const float* __restrict__ xl,
                         const float* __restrict__ eexp, const float* __restrict__ denom,
                         float* __restrict__ out, int H, int C) {
  const int F = H * C, nch = F >> 4;
  int i = blockIdx.x * blockDim.x + threadIdx.x;
  int total = (E + Nn) * nch;
  if (i >= total) return;
  int e = i / nch, ch = i - e * nch, s, d;
  edge_sd(ei, E, e, s, d);
  int h = (ch * 16) / C;
  float alpha = eexp[(size_t)e * H + h] / (denom[(size_t)d * H + h] + 1e-16f);
  const float4* src4 = (const float4*)(xl + (size_t)s * F + ch * 16);
  float* dst = out + (size_t)d * F + ch * 16;
  #pragma unroll
  for (int q = 0; q < 4; ++q) {
    float4 v = src4[q];
    atomicAdd(&dst[q * 4 + 0], v.x * alpha);
    atomicAdd(&dst[q * 4 + 1], v.y * alpha);
    atomicAdd(&dst[q * 4 + 2], v.z * alpha);
    atomicAdd(&dst[q * 4 + 3], v.w * alpha);
  }
}

// ---------------------------------------------------------------- epilogues
__global__ void elu_bias(float* __restrict__ acc, const float* __restrict__ b,
                         int Nn, int F) {
  int i = blockIdx.x * blockDim.x + threadIdx.x;
  if (i >= Nn * F) return;
  float v = acc[i] + b[i % F];
  acc[i] = v > 0.f ? v : expm1f(v);
}

__global__ void bias_add(float* __restrict__ out, const float* __restrict__ b,
                         int Nn, int F) {
  int i = blockIdx.x * blockDim.x + threadIdx.x;
  if (i >= Nn * F) return;
  out[i] += b[i % F];
}

// ---------------------------------------------------------------- launch
static inline unsigned gridFor(size_t n, unsigned blk) {
  return (unsigned)((n + blk - 1) / blk);
}

extern "C" void kernel_launch(void* const* d_in, const int* in_sizes, int n_in,
                              void* d_out, int out_size, void* d_ws, size_t ws_size,
                              hipStream_t stream) {
  const float* x   = (const float*)d_in[0];
  const int*   ei  = (const int*)  d_in[1];
  const float* W1  = (const float*)d_in[2];
  const float* as1 = (const float*)d_in[3];
  const float* ad1 = (const float*)d_in[4];
  const float* b1  = (const float*)d_in[5];
  const float* W2  = (const float*)d_in[6];
  const float* as2 = (const float*)d_in[7];
  const float* ad2 = (const float*)d_in[8];
  const float* b2  = (const float*)d_in[9];
  float* out = (float*)d_out;

  const int Nn = in_sizes[0] / 128;   // 100000
  const int E  = in_sizes[1] / 2;     // 1600000
  const int Et = E + Nn;              // edges + self-loops

  // workspace layout (floats)
  float* w = (float*)d_ws;
  size_t o = 0;
  float* xl1   = w + o; o += (size_t)Nn * 128;
  float* acc1  = w + o; o += (size_t)Nn * 128;   // becomes h after ELU
  float* asr1  = w + o; o += (size_t)Nn * 8;
  float* adt1  = w + o; o += (size_t)Nn * 8;
  float* emax1 = w + o; o += (size_t)Nn * 8;
  float* den1  = w + o; o += (size_t)Nn * 8;
  float* eexp1 = w + o; o += (size_t)Et * 8;
  float* xl2   = w + o; o += (size_t)Nn * 64;
  float* asr2  = w + o; o += (size_t)Nn;
  float* adt2  = w + o; o += (size_t)Nn;
  float* emax2 = w + o; o += (size_t)Nn;
  float* den2  = w + o; o += (size_t)Nn;
  float* eexp2 = w + o; o += (size_t)Et;

  const unsigned BLK = 256;
  const unsigned rowTiles = gridFor((size_t)Nn, 16);
  const unsigned gemmGrid = rowTiles < 1280u ? rowTiles : 1280u;  // persistent blocks

  // ---------------- Layer 1: GATConv(128 -> 8 heads x 16), concat ----------------
  gemm_wmma_f16<128, 128><<<gemmGrid, 256, 0, stream>>>(x, W1, xl1, Nn);
  attn_coef<<<gridFor((size_t)Nn * 8, BLK), BLK, 0, stream>>>(xl1, as1, ad1, asr1, adt1, Nn, 8, 16);

  fill_f32<<<gridFor((size_t)Nn * 8,   BLK), BLK, 0, stream>>>(emax1, -INFINITY, (size_t)Nn * 8);
  fill_f32<<<gridFor((size_t)Nn * 8,   BLK), BLK, 0, stream>>>(den1,  0.f,       (size_t)Nn * 8);
  fill_f32<<<gridFor((size_t)Nn * 128, BLK), BLK, 0, stream>>>(acc1,  0.f,       (size_t)Nn * 128);

  edge_max<<<gridFor((size_t)Et * 8, BLK), BLK, 0, stream>>>(ei, E, Nn, asr1, adt1, emax1, 8);
  edge_exp<<<gridFor((size_t)Et * 8, BLK), BLK, 0, stream>>>(ei, E, Nn, asr1, adt1, emax1, eexp1, den1, 8);
  edge_msg<<<gridFor((size_t)Et * 8, BLK), BLK, 0, stream>>>(ei, E, Nn, xl1, eexp1, den1, acc1, 8, 16);

  elu_bias<<<gridFor((size_t)Nn * 128, BLK), BLK, 0, stream>>>(acc1, b1, Nn, 128);

  // ---------------- Layer 2: GATConv(128 -> 1 head x 64), mean(=identity) --------
  gemm_wmma_f16<128, 64><<<gemmGrid, 128, 0, stream>>>(acc1, W2, xl2, Nn);
  attn_coef<<<gridFor((size_t)Nn, BLK), BLK, 0, stream>>>(xl2, as2, ad2, asr2, adt2, Nn, 1, 64);

  fill_f32<<<gridFor((size_t)Nn,      BLK), BLK, 0, stream>>>(emax2, -INFINITY, (size_t)Nn);
  fill_f32<<<gridFor((size_t)Nn,      BLK), BLK, 0, stream>>>(den2,  0.f,       (size_t)Nn);
  fill_f32<<<gridFor((size_t)Nn * 64, BLK), BLK, 0, stream>>>(out,   0.f,       (size_t)Nn * 64);

  edge_max<<<gridFor((size_t)Et, BLK), BLK, 0, stream>>>(ei, E, Nn, asr2, adt2, emax2, 1);
  edge_exp<<<gridFor((size_t)Et, BLK), BLK, 0, stream>>>(ei, E, Nn, asr2, adt2, emax2, eexp2, den2, 1);
  edge_msg<<<gridFor((size_t)Et * 4, BLK), BLK, 0, stream>>>(ei, E, Nn, xl2, eexp2, den2, out, 1, 64);

  bias_add<<<gridFor((size_t)Nn * 64, BLK), BLK, 0, stream>>>(out, b2, Nn, 64);
}
